// RelExLSTM_8323646620151
// MI455X (gfx1250) — matrix operand order repaired and checked by reading, not compiled, not run on hardware
//
#include <hip/hip_runtime.h>
#include <math.h>

// ---------- types ----------
typedef __attribute__((ext_vector_type(16))) __bf16        v16bf;
typedef __attribute__((ext_vector_type(8)))  float         v8f;
typedef __attribute__((ext_vector_type(4)))  unsigned int  u32x4;

union FragU { u32x4 q[2]; v16bf v; };   // 32B = one WMMA A/B fragment per lane

#define T_STEPS   100
#define BATCH_N   512
#define EMB_N     300
#define EMB_PAD   320            // padded K for x@W (multiple of 32)
#define H_N       256
#define G4H       1024           // 4*H
#define MTILE     32             // batch rows per workgroup
#define SCAN_THREADS 512         // 16 waves; wave w owns h-cols [16w,16w+16)

// ---------- helpers ----------
__device__ inline unsigned short f2bf(float f) {            // fp32 -> bf16 RNE
  unsigned int u = __float_as_uint(f);
  u += 0x7FFFu + ((u >> 16) & 1u);
  return (unsigned short)(u >> 16);
}
__device__ inline float sigm_(float x) { return 1.0f / (1.0f + __expf(-x)); }
__device__ inline float tanh_(float x) {
  x = fminf(fmaxf(x, -15.f), 15.f);
  float e = __expf(2.f * x);
  return (e - 1.f) / (e + 1.f);
}

// Issue one async memory->LDS b128 copy (CDNA5 Tensor/async path, ASYNCcnt-tracked).
// SADDR form: mem_addr = SGPR_base + VGPR_offset ; LDS dest = VDST VGPR (byte offset).
__device__ inline void async_b128(unsigned lds_byte_off, unsigned g_byte_off,
                                  const unsigned short* gbase) {
  asm volatile("global_load_async_to_lds_b128 %0, %1, %2"
               :: "v"(lds_byte_off), "v"(g_byte_off), "s"(gbase)
               : "memory");
}
__device__ inline void wait_async0() {
  asm volatile("s_wait_asynccnt 0x0" ::: "memory");
}

// ---------- prep: pack fp32 (K x 1024) row-major into WMMA-B fragment order ----------
// Fragment (kt, nt): lane L holds col = nt*16 + (L&15), K = kt*32 + (L>>4)*16 + j (j=0..15),
// stored as 32 contiguous bytes per lane -> perfectly coalesced b128 loads in the scan.
__global__ void pack_b_kernel(const float* __restrict__ src, int Ksrc, int Kt,
                              unsigned short* __restrict__ dst) {
  int tid = blockIdx.x * blockDim.x + threadIdx.x;
  int total = Kt * 64 * 32;
  if (tid >= total) return;
  int lane = tid & 31;
  int nt   = (tid >> 5) & 63;
  int kt   = tid >> 11;
  int col  = nt * 16 + (lane & 15);
  int kb   = kt * 32 + ((lane >> 4) << 4);
  alignas(16) unsigned short tmp[16];
#pragma unroll
  for (int j = 0; j < 16; ++j) {
    int k = kb + j;
    float v = (k < Ksrc) ? src[(size_t)k * G4H + col] : 0.f;
    tmp[j] = f2bf(v);
  }
  u32x4* d = (u32x4*)(dst + (size_t)tid * 16);
  d[0] = *(const u32x4*)&tmp[0];
  d[1] = *(const u32x4*)&tmp[8];
}

// ---------- prep: gather embeddings -> bf16 x[t][b][EMB_PAD] ----------
__global__ void gather_x_kernel(const int* __restrict__ tok, const float* __restrict__ emb,
                                unsigned short* __restrict__ xpk) {
  size_t tid = (size_t)blockIdx.x * blockDim.x + threadIdx.x;
  int e = (int)(tid % EMB_PAD);
  size_t bt = tid / EMB_PAD;
  int t = (int)(bt % T_STEPS);
  int b = (int)(bt / T_STEPS);
  if (b >= BATCH_N) return;
  float v = 0.f;
  if (e < EMB_N) {
    int token = tok[b * T_STEPS + t];
    v = emb[(size_t)token * EMB_N + e];
  }
  xpk[((size_t)t * BATCH_N + b) * EMB_PAD + e] = f2bf(v);
}

// ---------- persistent bidirectional LSTM scan ----------
// blockIdx.x = dir*16 + batch_tile.  h (bf16) lives in LDS, c lives in VGPRs for the
// whole 100-step scan.  z = x_t@W + h@U + b via v_wmma_f32_16x16x32_bf16; each wave
// owns 16 h-columns (all 4 gates).  x tiles are double-buffered in LDS and fetched
// with global_load_async_to_lds_b128 overlapping the WMMA phase (ASYNCcnt).
__global__ void __launch_bounds__(SCAN_THREADS, 1)
lstm_scan_kernel(const unsigned short* __restrict__ xpk,
                 const unsigned short* __restrict__ WpkF,
                 const unsigned short* __restrict__ WpkB,
                 const unsigned short* __restrict__ UpkF,
                 const unsigned short* __restrict__ UpkB,
                 const float* __restrict__ bf_,
                 const float* __restrict__ bb_,
                 float* __restrict__ hcat) {
  __shared__ alignas(16) unsigned short sX[2][MTILE * EMB_PAD];  // 2 x 20 KB
  __shared__ alignas(16) unsigned short sH[MTILE * H_N];         // 16 KB

  const int tid  = threadIdx.x;
  const int lane = tid & 31;
  const int w    = tid >> 5;                 // wave id 0..15
  const int dir  = blockIdx.x >> 4;
  const int mt0  = (blockIdx.x & 15) * MTILE;

  const unsigned short* Wpk  = dir ? WpkB : WpkF;
  const unsigned short* Upk  = dir ? UpkB : UpkF;
  const float*          bias = dir ? bb_  : bf_;

  float bgate[4];
#pragma unroll
  for (int g = 0; g < 4; ++g) bgate[g] = bias[g * H_N + w * 16 + (lane & 15)];

  for (int i = tid; i < MTILE * H_N; i += SCAN_THREADS) sH[i] = 0;

  v8f c[2];
#pragma unroll
  for (int m = 0; m < 2; ++m)
#pragma unroll
    for (int r = 0; r < 8; ++r) c[m][r] = 0.f;

  // ---- preload x tile for t=0 via async DMA to LDS ----
  {
    const int t0 = dir ? (T_STEPS - 1) : 0;
    const unsigned short* g0 = xpk + ((size_t)t0 * BATCH_N + mt0) * EMB_PAD;
    for (int i = tid; i < (MTILE * EMB_PAD) / 8; i += SCAN_THREADS)
      async_b128((unsigned)(size_t)&sX[0][i * 8], (unsigned)(i * 16), g0);
  }
  wait_async0();
  __syncthreads();

  for (int t = 0; t < T_STEPS; ++t) {
    const int buf = t & 1;
    const unsigned short* sXb = &sX[buf][0];

    // ---- kick off async DMA of next timestep's x tile into the other buffer ----
    if (t + 1 < T_STEPS) {
      const int tn = dir ? (T_STEPS - 2 - t) : (t + 1);
      const unsigned short* gn = xpk + ((size_t)tn * BATCH_N + mt0) * EMB_PAD;
      unsigned short* db = &sX[buf ^ 1][0];
      for (int i = tid; i < (MTILE * EMB_PAD) / 8; i += SCAN_THREADS)
        async_b128((unsigned)(size_t)&db[i * 8], (unsigned)(i * 16), gn);
    }
    if (t + 2 < T_STEPS) {  // warm L2 two steps ahead
      const int tp = dir ? (T_STEPS - 3 - t) : (t + 2);
      const char* pf = (const char*)(xpk + ((size_t)tp * BATCH_N + mt0) * EMB_PAD);
      __builtin_prefetch(pf + (size_t)tid * 40, 0, 0);
    }

    // accumulators = bias (per-column splat)
    v8f acc[4][2];
#pragma unroll
    for (int g = 0; g < 4; ++g)
#pragma unroll
      for (int m = 0; m < 2; ++m)
#pragma unroll
        for (int r = 0; r < 8; ++r) acc[g][m][r] = bgate[g];

    const int arow = lane & 15;
    const int koff = (lane >> 4) << 4;       // A-frag K sub-offset (contiguous, matches packed B)

    // ---- phase 1: z += x_t @ W  (K = 320, 10 k-tiles) ----
#pragma unroll 2
    for (int kt = 0; kt < 10; ++kt) {
      FragU a0, a1;
      const int k0 = kt * 32 + koff;
      a0.q[0] = *(const u32x4*)(sXb + (arow     ) * EMB_PAD + k0);
      a0.q[1] = *(const u32x4*)(sXb + (arow     ) * EMB_PAD + k0 + 8);
      a1.q[0] = *(const u32x4*)(sXb + (arow + 16) * EMB_PAD + k0);
      a1.q[1] = *(const u32x4*)(sXb + (arow + 16) * EMB_PAD + k0 + 8);
#pragma unroll
      for (int g = 0; g < 4; ++g) {
        const int ntile = g * 16 + w;
        const u32x4* bp = (const u32x4*)(Wpk + ((size_t)(kt * 64 + ntile) * 32 + lane) * 16);
        FragU bfr; bfr.q[0] = bp[0]; bfr.q[1] = bp[1];
        acc[g][0] = __builtin_amdgcn_wmma_f32_16x16x32_bf16(false, a0.v, false, bfr.v,
                                                            (short)0, acc[g][0], false, false);
        acc[g][1] = __builtin_amdgcn_wmma_f32_16x16x32_bf16(false, a1.v, false, bfr.v,
                                                            (short)0, acc[g][1], false, false);
      }
    }
    // ---- phase 2: z += h @ U  (K = 256, 8 k-tiles) ----
#pragma unroll 2
    for (int kt = 0; kt < 8; ++kt) {
      FragU a0, a1;
      const int k0 = kt * 32 + koff;
      a0.q[0] = *(const u32x4*)(sH + (arow     ) * H_N + k0);
      a0.q[1] = *(const u32x4*)(sH + (arow     ) * H_N + k0 + 8);
      a1.q[0] = *(const u32x4*)(sH + (arow + 16) * H_N + k0);
      a1.q[1] = *(const u32x4*)(sH + (arow + 16) * H_N + k0 + 8);
#pragma unroll
      for (int g = 0; g < 4; ++g) {
        const int ntile = g * 16 + w;
        const u32x4* bp = (const u32x4*)(Upk + ((size_t)(kt * 64 + ntile) * 32 + lane) * 16);
        FragU bfr; bfr.q[0] = bp[0]; bfr.q[1] = bp[1];
        acc[g][0] = __builtin_amdgcn_wmma_f32_16x16x32_bf16(false, a0.v, false, bfr.v,
                                                            (short)0, acc[g][0], false, false);
        acc[g][1] = __builtin_amdgcn_wmma_f32_16x16x32_bf16(false, a1.v, false, bfr.v,
                                                            (short)0, acc[g][1], false, false);
      }
    }
    __syncthreads();   // all waves finished reading sH for this step

    // ---- gates + state update; write new h (bf16) to LDS ----
#pragma unroll
    for (int m = 0; m < 2; ++m) {
#pragma unroll
      for (int r = 0; r < 8; ++r) {
        float iv = sigm_(acc[0][m][r]);
        float fv = sigm_(acc[1][m][r]);
        float gv = tanh_(acc[2][m][r]);
        float ov = sigm_(acc[3][m][r]);
        float cn = fv * c[m][r] + iv * gv;
        c[m][r]  = cn;
        float hv = ov * tanh_(cn);
        const int rrow = m * 16 + r + ((lane >> 4) << 3);   // C-layout row
        const int col  = w * 16 + (lane & 15);
        sH[rrow * H_N + col] = f2bf(hv);
        if (t == T_STEPS - 1)
          hcat[(size_t)(mt0 + rrow) * (2 * H_N) + dir * H_N + col] = hv;
      }
    }
    wait_async0();     // next x tile has landed in LDS (per-wave issues)
    __syncthreads();   // new h + next x tile visible to all waves
  }
}

// ---------- head: softmax(hcat @ Wd + bd) ----------
__global__ void head_kernel(const float* __restrict__ hcat, const float* __restrict__ Wd,
                            const float* __restrict__ bd, float* __restrict__ out) {
  __shared__ float sh[2 * H_N];
  __shared__ float slog[19];
  const int row = blockIdx.x;
  const int l = threadIdx.x;
  for (int k = l; k < 2 * H_N; k += 64) sh[k] = hcat[(size_t)row * (2 * H_N) + k];
  __syncthreads();
  if (l < 19) {
    float a = bd[l];
    for (int k = 0; k < 2 * H_N; ++k) a += sh[k] * Wd[k * 19 + l];
    slog[l] = a;
  }
  __syncthreads();
  if (l < 19) {
    float mx = -3.0e38f;
    for (int j = 0; j < 19; ++j) mx = fmaxf(mx, slog[j]);
    float s = 0.f;
    for (int j = 0; j < 19; ++j) s += __expf(slog[j] - mx);
    out[(size_t)row * 19 + l] = __expf(slog[l] - mx) / s;
  }
}

// ---------- launch ----------
extern "C" void kernel_launch(void* const* d_in, const int* in_sizes, int n_in,
                              void* d_out, int out_size, void* d_ws, size_t ws_size,
                              hipStream_t stream) {
  const int*   tok = (const int*)d_in[0];
  const float* emb = (const float*)d_in[1];
  const float* Wf  = (const float*)d_in[2];
  const float* Uf  = (const float*)d_in[3];
  const float* bf  = (const float*)d_in[4];
  const float* Wb  = (const float*)d_in[5];
  const float* Ub  = (const float*)d_in[6];
  const float* bb  = (const float*)d_in[7];
  const float* Wd  = (const float*)d_in[8];
  const float* bd  = (const float*)d_in[9];
  float* out = (float*)d_out;

  char* ws = (char*)d_ws;
  // workspace layout (bytes)
  unsigned short* WpkF = (unsigned short*)(ws + 0);          //  655,360
  unsigned short* WpkB = (unsigned short*)(ws + 655360);     //  655,360
  unsigned short* UpkF = (unsigned short*)(ws + 1310720);    //  524,288
  unsigned short* UpkB = (unsigned short*)(ws + 1835008);    //  524,288
  unsigned short* xpk  = (unsigned short*)(ws + 2359296);    // 32,768,000
  float*          hcat = (float*)(ws + 35127296);            //  1,048,576  (total ~36.2 MB)

  pack_b_kernel<<<80, 256, 0, stream>>>(Wf, EMB_N, 10, WpkF);
  pack_b_kernel<<<80, 256, 0, stream>>>(Wb, EMB_N, 10, WpkB);
  pack_b_kernel<<<64, 256, 0, stream>>>(Uf, H_N, 8, UpkF);
  pack_b_kernel<<<64, 256, 0, stream>>>(Ub, H_N, 8, UpkB);
  gather_x_kernel<<<64000, 256, 0, stream>>>(tok, emb, xpk);
  lstm_scan_kernel<<<32, SCAN_THREADS, 0, stream>>>(xpk, WpkF, WpkB, UpkF, UpkB,
                                                    bf, bb, hcat);
  head_kernel<<<512, 64, 0, stream>>>(hcat, Wd, bd, out);
}
